// SoftMOE_56281251447397
// MI455X (gfx1250) — compile-verified, matrix operand order checked by reading, not run on hardware
//
#include <hip/hip_runtime.h>
#include <hip/hip_bf16.h>
#include <math.h>

// ---------------------------------------------------------------------------
// Soft-MoE forward for MI455X (gfx1250, wave32, WMMA).
// B=4, L=1024, D=768, E=32, S=32, H=3072, K=E*S=1024.
// Weight-bandwidth bound (~604MB fp32 weights @ 23.3TB/s ~ 26us).
// All GEMM operands are laid out so WMMA fragments are two contiguous
// 16-byte runs per lane (A row-major [M,K], B stored transposed [N,K]),
// giving global_load_b128 fragment fetches feeding v_wmma_f32_16x16x32_bf16.
// Wave tile is 32x64: 2 A frags + 4 B frags -> 8 WMMA per K-step of 32
// (1.5 b128 loads per WMMA; each B fragment reused twice from registers).
// ---------------------------------------------------------------------------

typedef __attribute__((ext_vector_type(16))) __bf16       v16bf;
typedef __attribute__((ext_vector_type(8)))  float        v8f;
typedef __attribute__((ext_vector_type(4)))  unsigned int v4u;

#define NORM_EPS 1e-12f

__device__ __forceinline__ float warp_rsum(float v) {
#pragma unroll
  for (int m = 16; m >= 1; m >>= 1) v += __shfl_xor(v, m, 32);
  return v;
}
__device__ __forceinline__ float warp_rmax(float v) {
#pragma unroll
  for (int m = 16; m >= 1; m >>= 1) v = fmaxf(v, __shfl_xor(v, m, 32));
  return v;
}

// --------------------------- row L2-normalize x ----------------------------
// one wave per row (b,l); writes row-major xnb [B*L,D] (logits A matrix) and
// per-batch transposed xnbT [B][D][L] (slots B^T matrix).
__global__ void smoe_rownorm_x(const float* __restrict__ x,
                               __bf16* __restrict__ xnb,
                               __bf16* __restrict__ xnbT,
                               int Bn, int L, int D) {
  int row = blockIdx.x * blockDim.y + threadIdx.y;
  if (row >= Bn * L) return;
  int b = row / L, l = row % L;
  int lane = threadIdx.x;
  const float* xr = x + (size_t)row * D;
  float ss = 0.f;
  for (int d = lane; d < D; d += 32) { float v = xr[d]; ss += v * v; }
  ss = warp_rsum(ss);
  float inv = 1.f / fmaxf(sqrtf(ss), NORM_EPS);
  __bf16* o  = xnb + (size_t)row * D;
  __bf16* ot = xnbT + (size_t)b * D * L + l;
  for (int d = lane; d < D; d += 32) {
    __bf16 v = (__bf16)(xr[d] * inv);
    o[d] = v;
    ot[(size_t)d * L] = v;
  }
}

// ------------------- column L2-normalize phi (over D) ----------------------
// phi layout [D,K]; output stored TRANSPOSED: phinT [K,D] (coalesced writes),
// which is exactly the B^T layout the logits GEMM wants.
__global__ void smoe_colnorm_phi(const float* __restrict__ phi,
                                 const float* __restrict__ scale,
                                 __bf16* __restrict__ phinT, int D, int K) {
  int k = blockIdx.x * blockDim.y + threadIdx.y;
  if (k >= K) return;
  int lane = threadIdx.x;
  float ss = 0.f;
  for (int d = lane; d < D; d += 32) { float v = phi[(size_t)d * K + k]; ss += v * v; }
  ss = warp_rsum(ss);
  float inv = scale[0] / fmaxf(sqrtf(ss), NORM_EPS);
  __bf16* o = phinT + (size_t)k * D;
  for (int d = lane; d < D; d += 32) o[d] = (__bf16)(phi[(size_t)d * K + k] * inv);
}

// ------------------- fp32 -> bf16 transposing weight copy ------------------
// In: fp32 [Z][R][C] -> Out: bf16 [Z][C][R].  LDS 32x32 tile, both sides
// coalesced.  Used to pre-transpose w1/w2 so the MLP GEMMs get contiguous
// B^T fragments (HBM traffic neutral vs fused conversion, far fewer vmem ops).
__global__ void smoe_transpose_cvt(const float* __restrict__ In,
                                   __bf16* __restrict__ OutT, int R, int C) {
  __shared__ __bf16 tile[32][33];
  int z = blockIdx.z;
  const float* ip = In + (size_t)z * R * C;
  __bf16* op = OutT + (size_t)z * R * C;
  int c0 = blockIdx.x * 32, r0 = blockIdx.y * 32;
  int tx = threadIdx.x, ty = threadIdx.y;  // (32, 8)
#pragma unroll
  for (int i = 0; i < 4; i++) {
    int r = ty + i * 8;
    tile[r][tx] = (__bf16)ip[(size_t)(r0 + r) * C + c0 + tx];
  }
  __syncthreads();
#pragma unroll
  for (int i = 0; i < 4; i++) {
    int c = ty + i * 8;
    op[(size_t)(c0 + c) * R + r0 + tx] = tile[tx][c];
  }
}

// ------------------------------ dual softmax -------------------------------
// one wave per token row (b,l): lane e owns expert e's 32 contiguous logits.
// d_w  = softmax over S (per e)  -> stored transposed [B,E,S,L] (slots A)
// c_w  = softmax over K (global) -> stored [B,L,K]            (combine A)
__global__ void smoe_softmax(const float* __restrict__ logits,
                             __bf16* __restrict__ dwT,
                             __bf16* __restrict__ cw, int Bn, int L) {
  const int E = 32, S = 32, K = 1024;
  int row = blockIdx.x * blockDim.y + threadIdx.y;  // b*L + l
  if (row >= Bn * L) return;
  int b = row / L, l = row % L;
  int e = threadIdx.x;
  const float* lr = logits + (size_t)row * K + e * S;
  float xv[32];
  float me = -3.4e38f;
#pragma unroll
  for (int s = 0; s < S; s++) { xv[s] = lr[s]; me = fmaxf(me, xv[s]); }
  float se = 0.f;
#pragma unroll
  for (int s = 0; s < S; s++) { xv[s] = expf(xv[s] - me); se += xv[s]; }
  float inv_se = 1.f / se;
  __bf16* dwp = dwT + ((size_t)(b * E + e) * S) * L + l;
#pragma unroll
  for (int s = 0; s < S; s++) dwp[(size_t)s * L] = (__bf16)(xv[s] * inv_se);
  float gM = warp_rmax(me);
  float gS = warp_rsum(se * expf(me - gM));
  float f = expf(me - gM) / gS;
  __bf16* cwp = cw + (size_t)row * K + e * S;
#pragma unroll
  for (int s = 0; s < S; s++) cwp[s] = (__bf16)(xv[s] * f);
}

// ------------------------- generic WMMA GEMM -------------------------------
// C[z](M,N) = act(A[z](M,Kd) * B[z](Kd,N) + bias), fp32 accumulate.
// A: bf16 row-major [M,Kd].
// B: TB=__bf16 -> B^T bf16 [N,Kd] row-major (2 x b128 per fragment);
//    TB=float  -> B fp32 row-major [Kd,N], scalar loads + in-reg cvt (fallback).
// Per wave: 32x64 output tile: 2 A frags + 4 B frags -> 8 wmma per K-step.
// Fragment element j: k = (j/8)*16 + (lane<16?0:8) + (j%8) -> two 16B runs.
// C element: VGPR r -> M = m0+16*h+r(+8 hi half), N = n0+t*16+lane%16.
// Generic C addressing: off = (m/modm)*cm1 + (m%modm)*cm2 + col*cn  (allows
// transposed / scattered epilogue stores, e.g. eo^T for the combine GEMM).
// EPI: 0 none, 1 bias+exact GELU, 2 bias.
union SmoeFragU { v16bf v; v4u q[2]; unsigned u[8]; };
union SmoeBF2  { unsigned u; __bf16 h[2]; };

template <typename TB, typename TO, int EPI>
__global__ void smoe_gemm_wmma(const __bf16* __restrict__ Ab,
                               const TB* __restrict__ Bb, TO* __restrict__ Cb,
                               const float* __restrict__ bias,
                               int M, int N, int Kd, int lda, int ldb,
                               int modz, long sA1, long sA2, long sB1, long sB2,
                               long sC1, long sC2, long sbias,
                               int modm, long cm1, long cm2, long cn) {
  int z = blockIdx.z;
  int zr = z % modz, zq = z / modz;
  const __bf16* A = Ab + (size_t)zr * sA1 + (size_t)zq * sA2;
  const TB* Bp    = Bb + (size_t)zr * sB1 + (size_t)zq * sB2;
  TO* C           = Cb + (size_t)zr * sC1 + (size_t)zq * sC2;

  int lane = threadIdx.x;
  int m0 = (blockIdx.y * blockDim.y + threadIdx.y) * 32;
  if (m0 >= M) return;  // wave-uniform: EXEC stays all-ones for WMMA
  int n0 = blockIdx.x * 64;
  int kb = (lane < 16) ? 0 : 8;
  int nn = lane & 15;

  v8f acc[2][4];
  v8f vzero = {0.f, 0.f, 0.f, 0.f, 0.f, 0.f, 0.f, 0.f};
#pragma unroll
  for (int h = 0; h < 2; h++)
#pragma unroll
    for (int t = 0; t < 4; t++) acc[h][t] = vzero;

  const __bf16* arow0 = A + (size_t)(m0 + nn) * lda + kb;
  const __bf16* arow1 = A + (size_t)(m0 + 16 + nn) * lda + kb;
  const __bf16* brow[4];
  if constexpr (sizeof(TB) == 2) {
#pragma unroll
    for (int t = 0; t < 4; t++)
      brow[t] = (const __bf16*)Bp + (size_t)(n0 + t * 16 + nn) * ldb + kb;
  }

  for (int k0 = 0; k0 < Kd; k0 += 32) {
    SmoeFragU au0, au1;
    au0.q[0] = *reinterpret_cast<const v4u*>(arow0 + k0);
    au0.q[1] = *reinterpret_cast<const v4u*>(arow0 + k0 + 16);
    au1.q[0] = *reinterpret_cast<const v4u*>(arow1 + k0);
    au1.q[1] = *reinterpret_cast<const v4u*>(arow1 + k0 + 16);
    if constexpr (sizeof(TB) == 2) {
      if (k0 + 64 < Kd) {  // prefetch streaming operands 2 K-steps ahead
        __builtin_prefetch(arow0 + k0 + 64, 0, 3);
        __builtin_prefetch(arow1 + k0 + 64, 0, 3);
#pragma unroll
        for (int t = 0; t < 4; t++) __builtin_prefetch(brow[t] + k0 + 64, 0, 3);
      }
    }
#pragma unroll
    for (int t = 0; t < 4; t++) {
      SmoeFragU bu;
      if constexpr (sizeof(TB) == 2) {  // B^T bf16: contiguous fragment runs
        bu.q[0] = *reinterpret_cast<const v4u*>(brow[t] + k0);
        bu.q[1] = *reinterpret_cast<const v4u*>(brow[t] + k0 + 16);
      } else {  // fp32 row-major fallback: scalar gather + in-register cvt
        const TB* bcol = Bp + (n0 + t * 16 + nn);
#pragma unroll
        for (int i = 0; i < 8; i++) {
          int kk = k0 + kb + ((i >> 2) << 4) + ((i & 3) << 1);
          SmoeBF2 p;
          p.h[0] = (__bf16)(float)bcol[(size_t)kk * ldb];
          p.h[1] = (__bf16)(float)bcol[(size_t)(kk + 1) * ldb];
          bu.u[i] = p.u;
        }
      }
      acc[0][t] = __builtin_amdgcn_wmma_f32_16x16x32_bf16(
          false, au0.v, false, bu.v, (short)0, acc[0][t], false, false);
      acc[1][t] = __builtin_amdgcn_wmma_f32_16x16x32_bf16(
          false, au1.v, false, bu.v, (short)0, acc[1][t], false, false);
    }
  }

#pragma unroll
  for (int h = 0; h < 2; h++) {
#pragma unroll
    for (int t = 0; t < 4; t++) {
      int col = n0 + t * 16 + nn;
      float bv = 0.f;
      if constexpr (EPI != 0) bv = bias[(size_t)zr * sbias + col];
#pragma unroll
      for (int r = 0; r < 8; r++) {
        int mr = m0 + 16 * h + r + ((lane < 16) ? 0 : 8);
        float v = acc[h][t][r];
        if constexpr (EPI == 1) {  // bias + exact GELU
          v += bv;
          v = 0.5f * v * (1.f + erff(v * 0.70710678118654752f));
        }
        if constexpr (EPI == 2) v += bv;
        size_t off = (size_t)(mr / modm) * cm1 + (size_t)(mr % modm) * cm2 +
                     (size_t)col * cn;
        C[off] = (TO)v;
      }
    }
  }
}

// ---------------------------------------------------------------------------
extern "C" void kernel_launch(void* const* d_in, const int* in_sizes, int n_in,
                              void* d_out, int out_size, void* d_ws, size_t ws_size,
                              hipStream_t stream) {
  const int B = 4, L = 1024, D = 768, E = 32, S = 32, H = 3072, K = E * S;
  const float* x     = (const float*)d_in[0];
  const float* phi   = (const float*)d_in[1];
  const float* scale = (const float*)d_in[2];
  const float* w1    = (const float*)d_in[3];
  const float* b1    = (const float*)d_in[4];
  const float* w2    = (const float*)d_in[5];
  const float* b2    = (const float*)d_in[6];
  float* out = (float*)d_out;

  // --- workspace carve: ~86 MB core + 302 MB optional transposed weights ---
  char* w = (char*)d_ws;
  size_t used = 0;
  auto carve = [&](size_t bytes) {
    void* p = w;
    size_t pad = (bytes + 255) & ~(size_t)255;
    w += pad; used += pad;
    return p;
  };
  __bf16* xnb    = (__bf16*)carve((size_t)B * L * D * 2);     // [B*L, D]
  __bf16* xnbT   = (__bf16*)carve((size_t)B * D * L * 2);     // [B][D][L]
  __bf16* phinT  = (__bf16*)carve((size_t)K * D * 2);         // [K, D]
  float*  logits = (float*) carve((size_t)B * L * K * 4);     // [B*L, K]
  __bf16* dwT    = (__bf16*)carve((size_t)B * E * S * L * 2); // [B,E,S,L]
  __bf16* cw     = (__bf16*)carve((size_t)B * L * K * 2);     // [B*L, K]
  __bf16* slotsE = (__bf16*)carve((size_t)E * B * S * D * 2); // [E][B*S][D]
  __bf16* hE     = (__bf16*)carve((size_t)E * B * S * H * 2); // [E][B*S][H]
  __bf16* eoT    = (__bf16*)carve((size_t)B * D * K * 2);     // [B][D][K]
  __bf16* w1T = (__bf16*)carve((size_t)E * D * H * 2);        // [E][H][D]
  __bf16* w2T = (__bf16*)carve((size_t)E * H * D * 2);        // [E][D][H]
  bool bigws = (ws_size >= used);

  const int BIGM = 1 << 30;
  dim3 blk32x8(32, 8);

  // 1) normalizations (xn both layouts; phin transposed)
  smoe_rownorm_x<<<dim3((B * L + 7) / 8), blk32x8, 0, stream>>>(x, xnb, xnbT, B, L, D);
  smoe_colnorm_phi<<<dim3((K + 7) / 8), blk32x8, 0, stream>>>(phi, scale, phinT, D, K);

  // 2) logits[4096,1024] = xnb @ phinT^T (f32 out); 8 waves x 32 rows = 256 rows/block
  smoe_gemm_wmma<__bf16, float, 0><<<dim3(K / 64, (B * L) / 256, 1), blk32x8, 0, stream>>>(
      xnb, phinT, logits, nullptr, B * L, K, D, D, D,
      BIGM, 0, 0, 0, 0, 0, 0, 0, BIGM, 0, (long)K, 1);

  // 3) dual softmax
  smoe_softmax<<<dim3((B * L + 7) / 8), blk32x8, 0, stream>>>(logits, dwT, cw, B, L);

  // 4) slots: per z=(b*E+e): [32,768] = dwT[b,e]([S,L]) @ xnbT[b]^T -> slotsE[e][b*S..]
  smoe_gemm_wmma<__bf16, __bf16, 0><<<dim3(D / 64, 1, B * E), dim3(32, 1), 0, stream>>>(
      dwT, xnbT, slotsE, nullptr, S, D, L, L, L,
      E, (long)S * L, (long)E * S * L,       // A: e*(S*L) + b*(E*S*L)
      0, (long)D * L,                        // B^T: b*(D*L)
      (long)B * S * D, (long)S * D, 0,       // C: e*(B*S*D) + b*(S*D)
      BIGM, 0, (long)D, 1);

  if (bigws) {
    // pre-transpose+convert weights (HBM-neutral, fragment-contiguous GEMMs)
    smoe_transpose_cvt<<<dim3(H / 32, D / 32, E), blk32x8, 0, stream>>>(w1, w1T, D, H);
    smoe_transpose_cvt<<<dim3(D / 32, H / 32, E), blk32x8, 0, stream>>>(w2, w2T, H, D);

    // 5) h[e] = gelu(slotsE[e]([128,D]) @ w1T[e]^T + b1[e]); 4 waves = 128 rows
    smoe_gemm_wmma<__bf16, __bf16, 1><<<dim3(H / 64, 1, E), dim3(32, 4), 0, stream>>>(
        slotsE, w1T, hE, b1, B * S, H, D, D, D,
        BIGM, (long)B * S * D, 0, (long)H * D, 0, (long)B * S * H, 0, (long)H,
        BIGM, 0, (long)H, 1);

    // 6) eo^T: h[e]([128,H]) @ w2T[e]^T + b2[e], scattered TRANSPOSED into [B][D][K]
    //    row m=b*32+s, col=d -> eoT[b*D*K + d*K + e*S + s]
    smoe_gemm_wmma<__bf16, __bf16, 2><<<dim3(D / 64, 1, E), dim3(32, 4), 0, stream>>>(
        hE, w2T, eoT, b2, B * S, D, H, H, H,
        BIGM, (long)B * S * H, 0, (long)D * H, 0, (long)S, 0, (long)D,
        S, (long)D * K, 1, (long)K);
  } else {
    // fallback: fused fp32->bf16 conversion inside GEMM (scalar B gathers)
    smoe_gemm_wmma<float, __bf16, 1><<<dim3(H / 64, 1, E), dim3(32, 4), 0, stream>>>(
        slotsE, w1, hE, b1, B * S, H, D, D, H,
        BIGM, (long)B * S * D, 0, (long)D * H, 0, (long)B * S * H, 0, (long)H,
        BIGM, 0, (long)H, 1);
    smoe_gemm_wmma<float, __bf16, 2><<<dim3(D / 64, 1, E), dim3(32, 4), 0, stream>>>(
        hE, w2, eoT, b2, B * S, D, H, H, D,
        BIGM, (long)B * S * H, 0, (long)H * D, 0, (long)S, 0, (long)D,
        S, (long)D * K, 1, (long)K);
  }

  // 7) out[b]([L,D]) = cw[b]([L,K]) @ eoT[b]^T (f32 out); 256 rows/block
  smoe_gemm_wmma<__bf16, float, 0><<<dim3(D / 64, L / 256, B), blk32x8, 0, stream>>>(
      cw, eoT, out, nullptr, L, D, K, K, K,
      BIGM, (long)L * K, 0, (long)D * K, 0, (long)L * D, 0, 0,
      BIGM, 0, (long)D, 1);
}
